// GroupedQueryAttention_61297773249040
// MI455X (gfx1250) — matrix operand order repaired and checked by reading, not compiled, not run on hardware
//
#include <hip/hip_runtime.h>

typedef __attribute__((ext_vector_type(16))) __bf16 v16bf;
typedef __attribute__((ext_vector_type(8)))  float  v8f;
typedef __attribute__((ext_vector_type(4)))  unsigned int u32x4;
typedef __attribute__((ext_vector_type(4)))  int i32x4;
typedef __attribute__((ext_vector_type(8)))  int i32x8;

union FragBF { v16bf v; u32x4 q[2]; };

#define BS   2
#define SEQ  2048
#define DM   2048
#define NH   16
#define NKV  4
#define HD   128
#define MROWS (BS * SEQ)            // 4096
#define QKVN  (DM + 2 * NKV * HD)   // 3072 = 2048 q + 512 k + 512 v
#define ATT_SCALE 0.08838834764831845f   // 1/sqrt(128), folded into Q at RoPE
#define NEG_BIG  -1.0e9f

#if defined(__has_builtin)
#if __has_builtin(__builtin_amdgcn_global_load_async_to_lds_b128)
#define USE_ASYNC_LDS 1
#endif
#if __has_builtin(__builtin_amdgcn_tensor_load_to_lds) && \
    __has_builtin(__builtin_amdgcn_s_wait_tensorcnt)
#define USE_TDM 1
#endif
#endif

static __device__ __forceinline__ unsigned short f2bf(float f) {
  union { float f; unsigned u; } v; v.f = f;
  unsigned r = v.u + 0x7FFFu + ((v.u >> 16) & 1u);   // round-to-nearest-even
  return (unsigned short)(r >> 16);
}
static __device__ __forceinline__ v8f zero8() {
  return (v8f){0.f, 0.f, 0.f, 0.f, 0.f, 0.f, 0.f, 0.f};
}
static __device__ __forceinline__ u32x4 ld16g(const unsigned short* p) {
  return *(const u32x4*)p;
}

#ifdef USE_ASYNC_LDS
typedef __attribute__((address_space(1))) i32x4 as1_i32x4;
typedef __attribute__((address_space(3))) i32x4 as3_i32x4;

static __device__ __forceinline__ void cp_async_b128(const void* g, void* l) {
  __builtin_amdgcn_global_load_async_to_lds_b128(
      (as1_i32x4*)(unsigned long long)(size_t)g,
      (as3_i32x4*)(unsigned)(size_t)l,
      0, 0);
}
static __device__ __forceinline__ void wait_async0() {
#if __has_builtin(__builtin_amdgcn_s_wait_asynccnt)
  __builtin_amdgcn_s_wait_asynccnt(0);
#else
  asm volatile("s_wait_asynccnt 0x0" ::: "memory");
#endif
}
#endif

#ifdef USE_TDM
// Issue a TDM 2-D tile load: tile rows x cols (2-byte elems) from a row-major
// tensor with row stride = tensor_k elements, into LDS with 8-ushort row pad
// (pad_interval 32 DWORDs, pad_amount 4 DWORDs -> 72-ushort LDS rows).
// D# layout per CDNA5 ISA 8.3/8.4 (group0 128b, group1 256b; groups 2/3 = 0).
// Toolchain exposes the 6-arg builtin: (g0, g1, g2, g3, g4, cpol).
static __device__ __forceinline__ void tdm_load_2d(const unsigned short* gbase,
                                                   unsigned lds_off_bytes,
                                                   int tensor_k, int tensor_rows,
                                                   int tile_rows, int tile_cols) {
  const unsigned long long ga = (unsigned long long)(size_t)gbase;
  u32x4 g0;
  g0[0] = 1u;                                  // count=1 (valid user D#)
  g0[1] = lds_off_bytes;                       // lds_addr
  g0[2] = (unsigned)(ga & 0xFFFFFFFFu);        // global_addr[31:0]
  g0[3] = (unsigned)((ga >> 32) & 0x01FFFFFFu) // global_addr[56:32]
        | (2u << 30);                          // type = 2 ("image")

  i32x8 g1;
  g1[0] = (1 << 16)                            // data_size = 1 -> 2 bytes
        | (1 << 20)                            // pad_enable
        | (4 << 22)                            // pad_interval: 32 DWORDs
        | (3 << 25);                           // pad_amount: 4 DWORDs
  g1[1] = (tensor_k & 0xFFFF) << 16;                           // tensor_dim0[15:0]
  g1[2] = ((tensor_k >> 16) & 0xFFFF)                          // tensor_dim0[31:16]
        | ((tensor_rows & 0xFFFF) << 16);                      // tensor_dim1[15:0]
  g1[3] = ((tensor_rows >> 16) & 0xFFFF)                       // tensor_dim1[31:16]
        | ((tile_cols & 0xFFFF) << 16);                        // tile_dim0
  g1[4] = (tile_rows & 0xFFFF);                                // tile_dim1 (tile_dim2=0)
  g1[5] = tensor_k;                                            // dim0_stride[31:0]
  g1[6] = 0;                                                   // dim0_stride[47:32], dim1_stride[15:0]
  g1[7] = 0;                                                   // dim1_stride[47:16]

  const i32x4 z4 = {0, 0, 0, 0};
  const i32x8 z8 = {0, 0, 0, 0, 0, 0, 0, 0};
  __builtin_amdgcn_tensor_load_to_lds(g0, g1, z4, z4, z8, 0);
}
#endif

// ---------------------------------------------------------------------------
// fp32 -> bf16 convert (grid-stride)
// ---------------------------------------------------------------------------
__global__ void cvt_bf16_kernel(const float* __restrict__ src,
                                unsigned short* __restrict__ dst, int n) {
  for (int i = blockIdx.x * blockDim.x + threadIdx.x; i < n;
       i += gridDim.x * blockDim.x)
    dst[i] = f2bf(src[i]);
}

// Pack wq|wk|wv into one bf16 B matrix, TRANSPOSED to [QKVN][DM] (K-contig)
__global__ void pack_wqkv_t_kernel(const float* __restrict__ wq,
                                   const float* __restrict__ wk,
                                   const float* __restrict__ wv,
                                   unsigned short* __restrict__ dst) {
  const int total = DM * QKVN;
  for (int i = blockIdx.x * blockDim.x + threadIdx.x; i < total;
       i += gridDim.x * blockDim.x) {
    int k = i / QKVN;          // contraction index
    int c = i - k * QKVN;      // output column
    float v;
    if (c < DM)                 v = wq[(size_t)k * (NH * HD) + c];
    else if (c < DM + NKV * HD) v = wk[(size_t)k * (NKV * HD) + (c - DM)];
    else                        v = wv[(size_t)k * (NKV * HD) + (c - DM - NKV * HD)];
    dst[(size_t)c * DM + k] = f2bf(v);
  }
}

// Transpose + convert wo [K,N] fp32 -> [N][K] bf16
__global__ void cvt_transpose_kernel(const float* __restrict__ src,
                                     unsigned short* __restrict__ dst,
                                     int K, int N) {
  const int total = K * N;
  for (int i = blockIdx.x * blockDim.x + threadIdx.x; i < total;
       i += gridDim.x * blockDim.x) {
    int k = i / N;
    int c = i - k * N;
    dst[(size_t)c * K + k] = f2bf(src[(size_t)k * N + c]);
  }
}

// ---------------------------------------------------------------------------
// bf16 GEMM: C[M,N] = A[M,K] x Bt[N,K]^T, fp32 accumulate.
// Compile-time shapes so the epilogue uses immediate-offset stores.
// 256 threads = 8 waves; block tile 128(M) x 128(N); wave tile 32x64;
// k-step 64 (16 WMMA per staged tile per wave). Double-buffered staging:
// TDM descriptors when available, else async-b128, else plain b128.
// ---------------------------------------------------------------------------
template <int M, int N, int K>
__global__ __launch_bounds__(256)
void gemm_bf16_kernel(const unsigned short* __restrict__ A,
                      const unsigned short* __restrict__ Bt,
                      float* __restrict__ C) {
  __shared__ __align__(16) unsigned short As[2][128][72];  // [Mrow][K64] +pad
  __shared__ __align__(16) unsigned short Bs[2][128][72];  // [Ncol][K64] +pad

  const int tid  = threadIdx.x;
  const int lane = tid & 31;
  const int wave = tid >> 5;
  const int ln   = lane & 15;
  const int hi   = lane >> 4;
  const int wm   = wave & 3;    // 0..3 -> 32-row slice
  const int wn   = wave >> 2;   // 0..1 -> 64-col slice
  const int bm   = blockIdx.y * 128;
  const int bn   = blockIdx.x * 128;

  v8f acc[2][4];
  #pragma unroll
  for (int i = 0; i < 2; ++i)
    #pragma unroll
    for (int j = 0; j < 4; ++j) acc[i][j] = zero8();

  auto stage = [&](int buf, int k0) {
#ifdef USE_TDM
    if (wave == 0) {
      tdm_load_2d(A  + (size_t)bm * K + k0, (unsigned)(size_t)&As[buf][0][0],
                  K, M, 128, 64);
      tdm_load_2d(Bt + (size_t)bn * K + k0, (unsigned)(size_t)&Bs[buf][0][0],
                  K, N, 128, 64);
    }
#else
    #pragma unroll
    for (int i = 0; i < 4; ++i) {
      const int s   = tid + i * 256;   // 1024 b128 segments per matrix
      const int row = s >> 3;          // 0..127
      const int cs  = (s & 7) * 8;     // 0..56
      const unsigned short* ga = A  + (size_t)(bm + row) * K + k0 + cs;
      const unsigned short* gb = Bt + (size_t)(bn + row) * K + k0 + cs;
#ifdef USE_ASYNC_LDS
      cp_async_b128(ga, &As[buf][row][cs]);
      cp_async_b128(gb, &Bs[buf][row][cs]);
#else
      *(u32x4*)&As[buf][row][cs] = ld16g(ga);
      *(u32x4*)&Bs[buf][row][cs] = ld16g(gb);
#endif
    }
#endif
    if (k0 + 128 < K) {                // prefetch tile after next into L2
      const int prow = tid >> 1, pcs = (tid & 1) * 32;
      __builtin_prefetch(A  + (size_t)(bm + prow) * K + k0 + 128 + pcs, 0, 1);
      __builtin_prefetch(Bt + (size_t)(bn + prow) * K + k0 + 128 + pcs, 0, 1);
    }
  };

  auto stage_wait = [&]() {
#ifdef USE_TDM
    if (wave == 0) __builtin_amdgcn_s_wait_tensorcnt(0);
#elif defined(USE_ASYNC_LDS)
    wait_async0();
#endif
    __syncthreads();
  };

  stage(0, 0);
  stage_wait();

  int buf = 0;
  for (int k0 = 0; k0 < K; k0 += 64) {
    const int nk = k0 + 64;
    if (nk < K) stage(buf ^ 1, nk);

    #pragma unroll
    for (int kc = 0; kc < 2; ++kc) {
      FragBF a0, a1;
      const unsigned short* ar0 = &As[buf][wm * 32 + ln][kc * 32];
      const unsigned short* ar1 = &As[buf][wm * 32 + 16 + ln][kc * 32];
      a0.q[0] = *(const u32x4*)(ar0 + hi * 8);
      a0.q[1] = *(const u32x4*)(ar0 + 16 + hi * 8);
      a1.q[0] = *(const u32x4*)(ar1 + hi * 8);
      a1.q[1] = *(const u32x4*)(ar1 + 16 + hi * 8);

      #pragma unroll
      for (int j = 0; j < 4; ++j) {
        const unsigned short* br = &Bs[buf][wn * 64 + j * 16 + ln][kc * 32 + hi * 8];
        FragBF bf;
        bf.q[0] = *(const u32x4*)(br);
        bf.q[1] = *(const u32x4*)(br + 16);
        acc[0][j] = __builtin_amdgcn_wmma_f32_16x16x32_bf16(
            false, a0.v, false, bf.v, (short)0, acc[0][j], false, false);
        acc[1][j] = __builtin_amdgcn_wmma_f32_16x16x32_bf16(
            false, a1.v, false, bf.v, (short)0, acc[1][j], false, false);
      }
    }

    if (nk < K) {
      stage_wait();
      buf ^= 1;
    }
  }

  // Epilogue: constexpr strides -> immediate-offset stores off one base.
  float* cp = C + (size_t)(bm + wm * 32 + hi * 8) * N + bn + wn * 64 + ln;
  #pragma unroll
  for (int i = 0; i < 2; ++i)
    #pragma unroll
    for (int j = 0; j < 4; ++j)
      #pragma unroll
      for (int r = 0; r < 8; ++r)
        cp[(size_t)(i * 16 + r) * N + j * 16] = acc[i][j][r];
}

// ---------------------------------------------------------------------------
// RoPE + transpose + bf16 convert. Q is pre-scaled by 1/sqrt(HD).
// qkv fp32 [b*seq, 3072] -> qb [b,NH,s,HD], kb [b,NKV,s,HD],
//                           vt [b,NKV,HD,SEQ]  (V transposed: key-contig)
// ---------------------------------------------------------------------------
__global__ void rope_kernel(const float* __restrict__ qkv,
                            const float* __restrict__ fcos,
                            const float* __restrict__ fsin,
                            unsigned short* __restrict__ qb,
                            unsigned short* __restrict__ kb,
                            unsigned short* __restrict__ vt) {
  const int total = BS * NH * SEQ * (HD / 2);
  for (int idx = blockIdx.x * blockDim.x + threadIdx.x; idx < total;
       idx += gridDim.x * blockDim.x) {
    const int j  = idx & 63;          // pair index 0..63
    int t        = idx >> 6;
    const int s  = t & (SEQ - 1);
    t >>= 11;
    const int h  = t & (NH - 1);
    const int b  = t >> 4;

    const float* row = qkv + (size_t)(b * SEQ + s) * QKVN;
    const float cv = fcos[s * 64 + j];
    const float sv = fsin[s * 64 + j];

    const float q0 = row[h * HD + 2 * j];
    const float q1 = row[h * HD + 2 * j + 1];
    const size_t qo = ((size_t)(b * NH + h) * SEQ + s) * HD + 2 * j;
    qb[qo]     = f2bf((q0 * cv - q1 * sv) * ATT_SCALE);
    qb[qo + 1] = f2bf((q0 * sv + q1 * cv) * ATT_SCALE);

    if (h < NKV) {
      const float k0 = row[DM + h * HD + 2 * j];
      const float k1 = row[DM + h * HD + 2 * j + 1];
      const size_t ko = ((size_t)(b * NKV + h) * SEQ + s) * HD + 2 * j;
      kb[ko]     = f2bf(k0 * cv - k1 * sv);
      kb[ko + 1] = f2bf(k0 * sv + k1 * cv);

      const float v0 = row[DM + NKV * HD + h * HD + 2 * j];
      const float v1 = row[DM + NKV * HD + h * HD + 2 * j + 1];
      const size_t vo = ((size_t)(b * NKV + h) * HD + 2 * j) * SEQ + s;
      vt[vo]       = f2bf(v0);
      vt[vo + SEQ] = f2bf(v1);
    }
  }
}

// ---------------------------------------------------------------------------
// Flash attention: 1 wave / 16-row Q block. S=QK^T and O+=PV via WMMA bf16.
// Q pre-scaled; mask only applied on diagonal-straddling chunks.
// ---------------------------------------------------------------------------
__global__ __launch_bounds__(32)
void attn_kernel(const unsigned short* __restrict__ qb,
                 const unsigned short* __restrict__ kb,
                 const unsigned short* __restrict__ vt,
                 unsigned short* __restrict__ ob) {
  __shared__ __align__(16) unsigned short Psh[16][32];

  const int bid  = blockIdx.x;
  const int qblk = bid & 127;               // SEQ/16 = 128
  const int h    = (bid >> 7) & (NH - 1);
  const int b    = bid >> 11;
  const int kvh  = h >> 2;                  // rep = 4

  const int lane = threadIdx.x;
  const int ln   = lane & 15;
  const int hi   = lane >> 4;

  const unsigned short* Qb = qb + (size_t)(b * NH + h)    * SEQ * HD;
  const unsigned short* Kb = kb + (size_t)(b * NKV + kvh) * SEQ * HD;
  const unsigned short* Vt = vt + (size_t)(b * NKV + kvh) * HD * SEQ;

  // Q fragments: 16 rows x 128 feats = 4 A-frags, each 2 x b128 per lane
  const int qrow = qblk * 16 + ln;
  const unsigned short* qp = Qb + (size_t)qrow * HD;
  FragBF qf[4];
  #pragma unroll
  for (int kk = 0; kk < 4; ++kk) {
    qf[kk].q[0] = ld16g(qp + kk * 32 + hi * 8);
    qf[kk].q[1] = ld16g(qp + kk * 32 + 16 + hi * 8);
  }

  v8f o[8];
  #pragma unroll
  for (int nt = 0; nt < 8; ++nt) o[nt] = zero8();
  float m[8], l[8];
  #pragma unroll
  for (int r = 0; r < 8; ++r) { m[r] = -1.0e30f; l[r] = 0.0f; }

  const int keys_end = (qblk + 1) * 16;
  for (int k0 = 0; k0 < keys_end; k0 += 32) {
    if (k0 + 32 < keys_end) {
      __builtin_prefetch(Kb + (size_t)(k0 + 32 + ln) * HD, 0, 1);
      __builtin_prefetch(Vt + (size_t)ln * SEQ + k0 + 32, 0, 1);
    }
    const bool need_mask = (k0 + 31) > (qblk * 16);   // wave-uniform

    // ---- S = Q K^T for 32 keys (2 subtiles of 16) ----
    v8f s[2];
    #pragma unroll
    for (int n = 0; n < 2; ++n) {
      v8f c = zero8();
      const int key = k0 + n * 16 + ln;     // B col = key index
      const unsigned short* kp = Kb + (size_t)key * HD;
      #pragma unroll
      for (int kk = 0; kk < 4; ++kk) {
        FragBF kf;                          // table order: K = 8*hi + e (+16 for e>=8)
        kf.q[0] = ld16g(kp + kk * 32 + hi * 8);
        kf.q[1] = ld16g(kp + kk * 32 + 16 + hi * 8);
        c = __builtin_amdgcn_wmma_f32_16x16x32_bf16(false, qf[kk].v, false, kf.v,
                                                    (short)0, c, false, false);
      }
      if (need_mask) {
        #pragma unroll
        for (int r = 0; r < 8; ++r) {
          const int row = qblk * 16 + r + hi * 8;
          const int col = k0 + n * 16 + ln;
          if (col > row) c[r] += NEG_BIG;
        }
      }
      s[n] = c;
    }

    // ---- online softmax update (rows live in 16-lane halves) ----
    float f[8];
    #pragma unroll
    for (int r = 0; r < 8; ++r) {
      float cm = fmaxf(s[0][r], s[1][r]);
      cm = fmaxf(cm, __shfl_xor(cm, 1, 32));
      cm = fmaxf(cm, __shfl_xor(cm, 2, 32));
      cm = fmaxf(cm, __shfl_xor(cm, 4, 32));
      cm = fmaxf(cm, __shfl_xor(cm, 8, 32));
      const float nm = fmaxf(m[r], cm);
      f[r] = __expf(m[r] - nm);
      m[r] = nm;
      const float p0 = __expf(s[0][r] - nm);
      const float p1 = __expf(s[1][r] - nm);
      s[0][r] = p0; s[1][r] = p1;
      float rs = p0 + p1;
      rs += __shfl_xor(rs, 1, 32);
      rs += __shfl_xor(rs, 2, 32);
      rs += __shfl_xor(rs, 4, 32);
      rs += __shfl_xor(rs, 8, 32);
      l[r] = l[r] * f[r] + rs;
    }
    #pragma unroll
    for (int nt = 0; nt < 8; ++nt)
      #pragma unroll
      for (int r = 0; r < 8; ++r) o[nt][r] *= f[r];

    // ---- relayout P (C-frag -> A-frag) through LDS as bf16 ----
    #pragma unroll
    for (int n = 0; n < 2; ++n)
      #pragma unroll
      for (int r = 0; r < 8; ++r)
        Psh[r + hi * 8][n * 16 + ln] = f2bf(s[n][r]);
    __syncthreads();   // single-wave WG: lowers to S_NOP; DS ops in-order

    FragBF pf;         // A-frag read: row = ln, cols contiguous -> 2 x b128
    pf.q[0] = *(const u32x4*)&Psh[ln][hi * 8];
    pf.q[1] = *(const u32x4*)&Psh[ln][16 + hi * 8];
    __syncthreads();

    // ---- O += P V (8 output feature tiles of 16), V pre-transposed ----
    #pragma unroll
    for (int nt = 0; nt < 8; ++nt) {
      const unsigned short* vp = Vt + (size_t)(nt * 16 + ln) * SEQ + k0 + hi * 8;
      FragBF vf;
      vf.q[0] = ld16g(vp);
      vf.q[1] = ld16g(vp + 16);
      o[nt] = __builtin_amdgcn_wmma_f32_16x16x32_bf16(false, pf.v, false, vf.v,
                                                      (short)0, o[nt], false, false);
    }
  }

  // ---- normalize (1 rcp per row) & write O as bf16 ----
  float il[8];
  #pragma unroll
  for (int r = 0; r < 8; ++r) il[r] = __frcp_rn(l[r]);
  #pragma unroll
  for (int nt = 0; nt < 8; ++nt)
    #pragma unroll
    for (int r = 0; r < 8; ++r) {
      const float v = o[nt][r] * il[r];
      const int srow = qblk * 16 + r + hi * 8;
      ob[((size_t)(b * SEQ + srow) * DM) + h * HD + nt * 16 + ln] = f2bf(v);
    }
}

// ---------------------------------------------------------------------------
extern "C" void kernel_launch(void* const* d_in, const int* in_sizes, int n_in,
                              void* d_out, int out_size, void* d_ws, size_t ws_size,
                              hipStream_t stream) {
  const float* x    = (const float*)d_in[0];
  const float* fcos = (const float*)d_in[1];
  const float* fsin = (const float*)d_in[2];
  // d_in[3] = mask (unused; causal mask applied analytically)
  const float* wq   = (const float*)d_in[4];
  const float* wk   = (const float*)d_in[5];
  const float* wv   = (const float*)d_in[6];
  const float* wo   = (const float*)d_in[7];
  float* out        = (float*)d_out;

  char* ws = (char*)d_ws;
  size_t off = 0;
  auto alloc = [&](size_t bytes) -> void* {
    void* p = ws + off;
    off = (off + bytes + 255) & ~(size_t)255;
    return p;
  };
  unsigned short* xb    = (unsigned short*)alloc((size_t)MROWS * DM * 2);
  unsigned short* wqkvb = (unsigned short*)alloc((size_t)DM * QKVN * 2);   // [QKVN][DM]
  unsigned short* wob   = (unsigned short*)alloc((size_t)DM * DM * 2);     // [N][K]
  float*          qkv   = (float*)alloc((size_t)MROWS * QKVN * 4);
  unsigned short* qb    = (unsigned short*)alloc((size_t)BS * NH * SEQ * HD * 2);
  unsigned short* kb    = (unsigned short*)alloc((size_t)BS * NKV * SEQ * HD * 2);
  unsigned short* vt    = (unsigned short*)alloc((size_t)BS * NKV * HD * SEQ * 2);
  unsigned short* ob    = (unsigned short*)alloc((size_t)MROWS * DM * 2);

  // 1) precision conversion / weight packing (B operands pre-transposed)
  cvt_bf16_kernel<<<2048, 256, 0, stream>>>(x, xb, MROWS * DM);
  pack_wqkv_t_kernel<<<2048, 256, 0, stream>>>(wq, wk, wv, wqkvb);
  cvt_transpose_kernel<<<2048, 256, 0, stream>>>(wo, wob, DM, DM);

  // 2) fused QKV projection: [4096,2048] x [2048,3072]
  gemm_bf16_kernel<MROWS, QKVN, DM>
      <<<dim3(QKVN / 128, MROWS / 128), 256, 0, stream>>>(xb, wqkvb, qkv);

  // 3) RoPE + transpose to [b,h,s,d] bf16 (V stored key-contiguous)
  rope_kernel<<<4096, 256, 0, stream>>>(qkv, fcos, fsin, qb, kb, vt);

  // 4) flash attention, one wave per 16-row Q block
  attn_kernel<<<BS * NH * (SEQ / 16), 32, 0, stream>>>(qb, kb, vt, ob);

  // 5) output projection: [4096,2048] x [2048,2048] -> fp32 out
  gemm_bf16_kernel<MROWS, DM, DM>
      <<<dim3(DM / 128, MROWS / 128), 256, 0, stream>>>(ob, wob, out);
}
